// GCN_34626026340407
// MI455X (gfx1250) — matrix operand order repaired and checked by reading, not compiled, not run on hardware
//
#include <hip/hip_runtime.h>
#include <hip/hip_bf16.h>

typedef __attribute__((ext_vector_type(16))) _Float16 v16h;
typedef __attribute__((ext_vector_type(4)))  _Float16 v4h;
typedef __attribute__((ext_vector_type(2)))  _Float16 v2h;
typedef __attribute__((ext_vector_type(8)))  float    v8f;

// ---------------------------------------------------------------------------
// WMMA GEMM: C[M x N] = A[M x K] * B[K x N], A/B f32 in memory, f16 MACs with
// f32 accumulation via V_WMMA_F32_16X16X32_F16, C written as f16 (it only
// feeds the SpMM gather, which multiplies in f32). One wave per 16-col tile,
// RT row-tiles (16 rows each) per block, K consumed in 32-deep LDS slabs.
// RT=10 / 160 rows per block makes every staging loop an exact multiple of
// the thread count for both instantiations (no predicated tails), and lets
// one B slab feed 10 WMMAs per wave.
//
// LDS layout: A slab (row-major, 16 rows x 32 k per tile) and B slab
// (TRANSPOSED: N cols x 32 k) use an 80-byte (40-half) row stride so each
// lane's two ds_load_b128 fragment reads start at bank (row*20) mod 64 —
// 16 distinct start banks x 4 banks each = all 64 banks, conflict-free.
// ---------------------------------------------------------------------------
template <int K, int N, int RT>
__global__ __launch_bounds__((N / 16) * 32)
void gemm_wmma_f16(const float* __restrict__ A, const float* __restrict__ B,
                   _Float16* __restrict__ C, int M) {
  constexpr int WAVES   = N / 16;
  constexpr int THREADS = WAVES * 32;
  constexpr int LDP     = 40;  // padded row stride in halves (80 B)
  constexpr int A4      = RT * 16 * 8;   // float4s in the A slab
  constexpr int AIT     = A4 / THREADS;  // exact (RT=10: 5 @256thr, 10 @128thr)
  constexpr int B4      = 32 * N / 4;    // float4s in the B slab
  constexpr int BIT     = B4 / THREADS;  // exact (4 for both configs)
  static_assert(AIT * THREADS == A4 && BIT * THREADS == B4, "exact staging");

  __shared__ _Float16 sA[RT * 16 * LDP];  // RT*16 rows x 32 k (padded)
  __shared__ _Float16 sBt[N * LDP];       // N cols x 32 k (transposed, padded)

  const int tid  = threadIdx.x;
  const int lane = tid & 31;
  const int wv   = tid >> 5;          // which 16-wide N tile this wave owns
  const int half = lane >> 4;         // lane group 0: lanes 0-15, 1: 16-31
  const int mrow = lane & 15;

  const int rowBlk = blockIdx.x * (RT * 16);

  v8f acc[RT];
#pragma unroll
  for (int t = 0; t < RT; ++t) acc[t] = v8f{};

  for (int kt = 0; kt < K; kt += 32) {
    __syncthreads();
    // ---- Stage A slab: batch all b128 loads first (one clause), then
    //      convert + b64 LDS stores. No tails, no per-chunk waits.
    float4 areg[AIT];
#pragma unroll
    for (int it = 0; it < AIT; ++it) {
      const int idx = tid + it * THREADS;
      const int r = idx >> 3, c4 = idx & 7;
      areg[it] = *reinterpret_cast<const float4*>(
          &A[(size_t)(rowBlk + r) * K + kt + c4 * 4]);
    }
#pragma unroll
    for (int it = 0; it < AIT; ++it) {
      const int idx = tid + it * THREADS;
      const int r = idx >> 3, c4 = idx & 7;
      const v4h hv = {(_Float16)areg[it].x, (_Float16)areg[it].y,
                      (_Float16)areg[it].z, (_Float16)areg[it].w};
      *reinterpret_cast<v4h*>(&sA[r * LDP + c4 * 4]) = hv;
    }
    // ---- Stage B slab transposed: batch loads, then strided b16 stores
    //      (the transpose cost, amortized over RT*WAVES fragment reads).
    float4 breg[BIT];
#pragma unroll
    for (int it = 0; it < BIT; ++it) {
      const int idx = tid + it * THREADS;
      const int r = idx / (N / 4), c4 = idx % (N / 4);
      breg[it] = *reinterpret_cast<const float4*>(
          &B[(size_t)(kt + r) * N + c4 * 4]);
    }
#pragma unroll
    for (int it = 0; it < BIT; ++it) {
      const int idx = tid + it * THREADS;
      const int r = idx / (N / 4), c4 = idx % (N / 4);
      sBt[(c4 * 4 + 0) * LDP + r] = (_Float16)breg[it].x;
      sBt[(c4 * 4 + 1) * LDP + r] = (_Float16)breg[it].y;
      sBt[(c4 * 4 + 2) * LDP + r] = (_Float16)breg[it].z;
      sBt[(c4 * 4 + 3) * LDP + r] = (_Float16)breg[it].w;
    }
    __syncthreads();

    // B fragment (shared across all RT row tiles this k-step).
    // Layout (ISA 7.12.2, 16-bit B): lane-half h holds K = 8h..8h+7 and
    // 16+8h..16+8h+7 for column N = wv*16 + (lane&15). Two contiguous 8-half
    // chunks in sBt -> 2x ds_load_b128.
    v16h bf;
#pragma unroll
    for (int e = 0; e < 16; ++e) {
      bf[e] = sBt[(wv * 16 + mrow) * LDP + half * 8 + (e < 8 ? e : e + 8)];
    }

#pragma unroll
    for (int t = 0; t < RT; ++t) {
      // A fragment. Layout (ISA 7.12.2, 16-bit A 16x32):
      //   vgpr v (=e>>1), pair lo (=e&1):
      //   K = (v<4 ? 2v : 16+2(v-4)) + 8*half + lo,  M = lane&15.
      // For fixed lane this is two contiguous 8-half chunks -> 2x b128.
      v16h af;
#pragma unroll
      for (int e = 0; e < 16; ++e) {
        const int v  = e >> 1;
        const int lo = e & 1;
        const int kk = (v < 4 ? 2 * v : 16 + 2 * (v - 4)) + 8 * half + lo;
        af[e] = sA[(t * 16 + mrow) * LDP + kk];
      }
      acc[t] = __builtin_amdgcn_wmma_f32_16x16x32_f16(
          /*neg_a=*/false, af, /*neg_b=*/false, bf,
          /*c_mod=*/(short)0, acc[t], /*reuse_a=*/false, /*reuse_b=*/false);
    }
  }

  // Write back as f16: C/D layout — vgpr r: M = r + 8*half, N = lane&15.
#pragma unroll
  for (int t = 0; t < RT; ++t) {
    const int row0 = rowBlk + t * 16;
#pragma unroll
    for (int r = 0; r < 8; ++r) {
      const int m = r + 8 * half;
      C[(size_t)(row0 + m) * N + wv * 16 + mrow] = (_Float16)acc[t][r];
    }
  }
}

// ---------------------------------------------------------------------------
// SpMM scatter: out[dst[e]] += vals[e] * h[src[e]], one wave32 per edge.
// h is f16 (halves the L2-resident gather traffic); each lane gathers D/32
// contiguous halves and accumulates with f32 global atomics. Edge metadata is
// forced wave-uniform via readfirstlane so it comes in on the scalar path.
// ---------------------------------------------------------------------------
template <int D>
__global__ __launch_bounds__(256)
void spmm_atomic(const _Float16* __restrict__ h, const int* __restrict__ src,
                 const int* __restrict__ dst, const float* __restrict__ vals,
                 float* __restrict__ out, int E) {
  constexpr int VPL = D / 32;  // elements per lane
  const int lane = threadIdx.x & 31;
  const int nw   = gridDim.x * (blockDim.x >> 5);
  int e = blockIdx.x * (blockDim.x >> 5) + (threadIdx.x >> 5);

  for (; e < E; e += nw) {
    const int eu = __builtin_amdgcn_readfirstlane(e);  // wave-uniform -> SMEM
    if (eu + nw < E) {  // pull the next edge-list lines toward us early
      __builtin_prefetch(&src[eu + nw], 0, 1);
      __builtin_prefetch(&dst[eu + nw], 0, 1);
      __builtin_prefetch(&vals[eu + nw], 0, 1);
    }
    const int   s  = src[eu];
    const int   d0 = dst[eu];
    const float v  = vals[eu];
    const _Float16* hp = h + (size_t)s * D + lane * VPL;
    float*          op = out + (size_t)d0 * D + lane * VPL;
    if constexpr (VPL == 4) {
      const v4h hv = *reinterpret_cast<const v4h*>(hp);
      atomicAdd(op + 0, v * (float)hv[0]);
      atomicAdd(op + 1, v * (float)hv[1]);
      atomicAdd(op + 2, v * (float)hv[2]);
      atomicAdd(op + 3, v * (float)hv[3]);
    } else {
      const v2h hv = *reinterpret_cast<const v2h*>(hp);
      atomicAdd(op + 0, v * (float)hv[0]);
      atomicAdd(op + 1, v * (float)hv[1]);
    }
  }
}

// h[i, j] = relu(h[i, j] + b[j]) for a 128-wide f32 matrix, float4/thread.
__global__ __launch_bounds__(256)
void bias_relu128(float* __restrict__ h, const float* __restrict__ b, int rows) {
  const int i = blockIdx.x * blockDim.x + threadIdx.x;  // index in float4 units
  if (i >= rows * 32) return;
  const int j = (i & 31) * 4;  // column base within the 128-wide row
  float4* p = reinterpret_cast<float4*>(h) + i;
  float4 t = *p;
  t.x = fmaxf(t.x + b[j + 0], 0.f);
  t.y = fmaxf(t.y + b[j + 1], 0.f);
  t.z = fmaxf(t.z + b[j + 2], 0.f);
  t.w = fmaxf(t.w + b[j + 3], 0.f);
  *p = t;
}

// In-place: out[row] = log_softmax(out[row] + b2), 64 cols, one wave per row.
__global__ __launch_bounds__(256)
void bias_logsoftmax64(float* __restrict__ out, const float* __restrict__ b2,
                       int rows) {
  const int lane = threadIdx.x & 31;
  const int row  = blockIdx.x * (blockDim.x >> 5) + (threadIdx.x >> 5);
  if (row >= rows) return;
  float a0 = out[(size_t)row * 64 + lane]      + b2[lane];
  float a1 = out[(size_t)row * 64 + 32 + lane] + b2[32 + lane];

  float m = fmaxf(a0, a1);
#pragma unroll
  for (int off = 16; off > 0; off >>= 1) m = fmaxf(m, __shfl_xor(m, off, 32));

  float s = __expf(a0 - m) + __expf(a1 - m);
#pragma unroll
  for (int off = 16; off > 0; off >>= 1) s += __shfl_xor(s, off, 32);

  const float lse = m + __logf(s);
  out[(size_t)row * 64 + lane]      = a0 - lse;
  out[(size_t)row * 64 + 32 + lane] = a1 - lse;
}

// ---------------------------------------------------------------------------
// Launch: x @ W1 -> spmm -> +b1,relu -> @ W2 -> spmm(into d_out) -> logsoftmax
// ws layout: [h0 : NN*128 f16][h1 : NN*128 f32]; h2 (f16) reuses h0's region.
// ---------------------------------------------------------------------------
extern "C" void kernel_launch(void* const* d_in, const int* in_sizes, int n_in,
                              void* d_out, int out_size, void* d_ws, size_t ws_size,
                              hipStream_t stream) {
  const float* x    = (const float*)d_in[0];
  const int*   src  = (const int*)d_in[1];
  const int*   dst  = (const int*)d_in[2];
  const float* vals = (const float*)d_in[3];
  const float* W1   = (const float*)d_in[4];
  const float* b1   = (const float*)d_in[5];
  const float* W2   = (const float*)d_in[6];
  const float* b2   = (const float*)d_in[7];
  float* out = (float*)d_out;

  const int NN = in_sizes[0] / 256;  // 100000
  const int E  = in_sizes[1];        // 1600000

  _Float16* h0 = (_Float16*)d_ws;                  // NN x 128 f16
  float*    h1 = (float*)(h0 + (size_t)NN * 128);  // NN x 128 f32 accumulator
  _Float16* h2 = h0;                               // NN x 64 f16, reuses h0

  // Zero the two scatter accumulators every call (graph-replay safe).
  hipMemsetAsync(h1, 0, (size_t)NN * 128 * sizeof(float), stream);
  hipMemsetAsync(d_out, 0, (size_t)NN * 64 * sizeof(float), stream);

  // Layer 1 GEMM: 625 blocks x 8 waves, 160 rows/block (100000 = 625*160).
  gemm_wmma_f16<256, 128, 10><<<NN / 160, 256, 0, stream>>>(x, W1, h0, NN);
  spmm_atomic<128><<<8192, 256, 0, stream>>>(h0, src, dst, vals, h1, E);
  bias_relu128<<<(NN * 32 + 255) / 256, 256, 0, stream>>>(h1, b1, NN);

  // Layer 2 GEMM: 625 blocks x 4 waves.
  gemm_wmma_f16<128, 64, 10><<<NN / 160, 128, 0, stream>>>(h1, W2, h2, NN);
  spmm_atomic<64><<<8192, 256, 0, stream>>>(h2, src, dst, vals, out, E);
  bias_logsoftmax64<<<(NN + 7) / 8, 256, 0, stream>>>(out, b2, NN);
}
